// DeformableTransformerEncoderLayer_20048907337747
// MI455X (gfx1250) — matrix operand order repaired and checked by reading, not compiled, not run on hardware
//
#include <hip/hip_runtime.h>
#include <hip/hip_bf16.h>

typedef __attribute__((ext_vector_type(16))) _Float16 v16h;
typedef __attribute__((ext_vector_type(8)))  _Float16 v8h;
typedef __attribute__((ext_vector_type(8)))  float    v8f;

// Async global->LDS path (CDNA5): gate on device pass + builtin availability.
#if defined(__gfx1250__) && __has_builtin(__builtin_amdgcn_global_load_async_to_lds_b128) && __has_builtin(__builtin_amdgcn_s_wait_asynccnt)
#define USE_ASYNC_LDS 1
typedef int v4i_t __attribute__((ext_vector_type(4)));
typedef __attribute__((address_space(1))) v4i_t* as1_v4i;
typedef __attribute__((address_space(3))) v4i_t* as3_v4i;
#else
#define USE_ASYNC_LDS 0
#endif

// ---------------- constants (from the reference) ----------------
// B=4, L=11253, N=B*L=45012, DM=256, M=8, DH=32, LV=4, P=4, DF=1024.
// Npad = 45056 = 352*128 so WMMA tiles need no guards.

// ---------------- prep: f16 convert + query fuse + pad-zero ----------------
__global__ void prep_kernel(const float* __restrict__ src,
                            const float* __restrict__ q_feat,
                            const float* __restrict__ q_pos,
                            _Float16* __restrict__ src16,
                            _Float16* __restrict__ q16,
                            _Float16* __restrict__ samp16,
                            int N)
{
    size_t i = (size_t)blockIdx.x * blockDim.x + threadIdx.x;
    int row = (int)(i >> 8);
    if (row < N) {
        src16[i] = (_Float16)src[i];
        q16[i]   = (_Float16)(q_feat[i] + q_pos[i]);
    } else {
        src16[i]  = (_Float16)0.f;
        q16[i]    = (_Float16)0.f;
        samp16[i] = (_Float16)0.f;   // pad rows of sampled output stay zero
    }
}

// ---------------- weight transpose + f16 convert: Wt[c*K+k] = W[k*Nc+c] ----
__global__ void wtrans_kernel(const float* __restrict__ W,
                              _Float16* __restrict__ Wt, int K, int Nc)
{
    int i = blockIdx.x * blockDim.x + threadIdx.x;
    if (i >= K * Nc) return;
    int c = i / K, k = i - c * K;
    Wt[i] = (_Float16)W[(size_t)k * Nc + c];
}

// ---------------- WMMA GEMM with LDS-staged B ------------------------------
// C = A(f16,[Npad,K]) x B + bias.  Bt is column-major f16 ([Nc, K]).
// 8 waves/block: 128 rows x 64 cols.  The 64-col x K weight slice is staged
// through LDS once per block (8x less B traffic), double-buffered in K-chunks
// of 64 via GLOBAL_LOAD_ASYNC_TO_LDS_B128 (ASYNCcnt) when available.
#define KC 64
#define LDS_STRIDE 72   // KC + 8 halfs: 16 fragment columns start on 16 banks

__global__ __launch_bounds__(256)
void wmma_gemm_kernel(const _Float16* __restrict__ A,
                      const _Float16* __restrict__ Bt,
                      const float* __restrict__ bias,
                      float* __restrict__ Cf,          // may be null
                      _Float16* __restrict__ Ch,       // may be null
                      int K, int Nc, int relu)
{
    __shared__ _Float16 ldsB[2][64 * LDS_STRIDE];      // 18 KB

    const int lane = threadIdx.x & 31;
    const int wave = threadIdx.x >> 5;
    const int r    = lane & 15;     // row-within-tile for A, col for B/D
    const int hi   = lane >> 4;     // half-wave selector

    const int row0 = blockIdx.x * 128 + wave * 16;
    const int col0 = blockIdx.y * 64;

    v8f acc[4];
#pragma unroll
    for (int t = 0; t < 4; ++t)
#pragma unroll
        for (int e = 0; e < 8; ++e) acc[t][e] = 0.f;

    // A layout (16-bit 16x32): lane hi=0 -> K {0..7,16..23}, hi=1 -> +8
    const _Float16* aRow = A + (size_t)(row0 + r) * K + hi * 8;

    // Cooperative stage of one 64-col x KC chunk of B into LDS.
    auto stage = [&](int buf, int k0) {
        const int t = threadIdx.x;
#pragma unroll
        for (int pass = 0; pass < 2; ++pass) {
            int seg = t + pass * 256;          // 512 segments of 8 halfs
            int col = seg >> 3;                // 0..63
            int s8  = (seg & 7) * 8;           // half offset within chunk
            const _Float16* gp = Bt + (size_t)(col0 + col) * K + k0 + s8;
            _Float16* lp = &ldsB[buf][col * LDS_STRIDE + s8];
#if USE_ASYNC_LDS
            __builtin_amdgcn_global_load_async_to_lds_b128(
                (as1_v4i)gp, (as3_v4i)lp, 0, 0);
#else
            *(v8h*)lp = *(const v8h*)gp;
#endif
        }
    };

    stage(0, 0);
#if USE_ASYNC_LDS
    __builtin_amdgcn_s_wait_asynccnt(0);
#endif
    __syncthreads();

    const int nch = K / KC;
    for (int c = 0; c < nch; ++c) {
        if (c + 1 < nch) stage((c + 1) & 1, (c + 1) * KC);

        const _Float16* lb = ldsB[c & 1];
        const int kbase = c * KC;
#pragma unroll
        for (int kk = 0; kk < KC; kk += 32) {
            v8h alo = *(const v8h*)(aRow + kbase + kk);
            v8h ahi = *(const v8h*)(aRow + kbase + kk + 16);
            v16h a;
#pragma unroll
            for (int i = 0; i < 8; ++i) { a[i] = alo[i]; a[i + 8] = ahi[i]; }
#pragma unroll
            for (int t4 = 0; t4 < 4; ++t4) {
                // B layout (32x16): lane = col (r), hi=0 -> K {0..7,16..23}, hi=1 -> +8
                const _Float16* bp = lb + (t4 * 16 + r) * LDS_STRIDE + hi * 8 + kk;
                v8h blo = *(const v8h*)bp;
                v8h bhi = *(const v8h*)(bp + 16);
                v16h b;
#pragma unroll
                for (int i = 0; i < 8; ++i) { b[i] = blo[i]; b[i + 8] = bhi[i]; }
                acc[t4] = __builtin_amdgcn_wmma_f32_16x16x32_f16(
                    false, a, false, b, (short)0, acc[t4], false, false);
            }
        }
#if USE_ASYNC_LDS
        __builtin_amdgcn_s_wait_asynccnt(0);
#endif
        __syncthreads();
    }

    // D layout: lane = col (r), VGPR v = row v + 8*hi
#pragma unroll
    for (int t = 0; t < 4; ++t) {
        int col = col0 + t * 16 + r;
        float bs = bias ? bias[col] : 0.f;
#pragma unroll
        for (int v = 0; v < 8; ++v) {
            int rowm = row0 + v + 8 * hi;
            float x = acc[t][v] + bs;
            if (relu) x = fmaxf(x, 0.f);
            size_t idx = (size_t)rowm * Nc + col;
            if (Cf) Cf[idx] = x;
            if (Ch) Ch[idx] = (_Float16)x;
        }
    }
}

// ---------------- softmax over LV*P = 16, one thread per (n, head) --------
__global__ void softmax_kernel(float* __restrict__ attn, int N)
{
    int g = blockIdx.x * blockDim.x + threadIdx.x;
    if (g >= N * 8) return;
    float* p = attn + (size_t)(g >> 3) * 128 + (size_t)(g & 7) * 16;
    float v[16], mx = -3.4e38f;
#pragma unroll
    for (int i = 0; i < 16; ++i) { v[i] = p[i]; mx = fmaxf(mx, v[i]); }
    float s = 0.f;
#pragma unroll
    for (int i = 0; i < 16; ++i) { v[i] = __expf(v[i] - mx); s += v[i]; }
    float inv = 1.f / s;
#pragma unroll
    for (int i = 0; i < 16; ++i) p[i] = v[i] * inv;
}

// ---------------- deformable sampling: one wave per (n, head), lane = d ----
__global__ __launch_bounds__(256)
void sample_kernel(const float* __restrict__ value,   // [N,256] = (B,L,M,DH)
                   const float* __restrict__ off,     // [N,256] = (M,LV,P,2)
                   const float* __restrict__ attw,    // [N,128] = (M,LV*P)
                   const float* __restrict__ ref,     // [N,4,2]
                   _Float16* __restrict__ samp,       // [Npad,256]
                   int Lseq)
{
    const int n    = blockIdx.x;
    const int m    = threadIdx.x >> 5;
    const int lane = threadIdx.x & 31;
    const int b    = n / Lseq;

    const int HW[4] = {92, 46, 23, 12};
    const int S0[4] = {0, 8464, 10580, 11109};

    const size_t vbase = (size_t)b * Lseq * 256 + (size_t)m * 32 + lane;
    float acc = 0.f;
#pragma unroll
    for (int l = 0; l < 4; ++l) {
        const int H = HW[l], W = HW[l];
        const size_t lbase = vbase + (size_t)S0[l] * 256;
        float rx = ref[((size_t)n * 4 + l) * 2 + 0];
        float ry = ref[((size_t)n * 4 + l) * 2 + 1];
#pragma unroll
        for (int p = 0; p < 4; ++p) {
            int oi = ((m * 4 + l) * 4 + p) * 2;
            // loc = ref + off/(W,H); x = loc_x*W - 0.5 == rx*W + offx - 0.5
            float x = rx * (float)W + off[(size_t)n * 256 + oi]     - 0.5f;
            float y = ry * (float)H + off[(size_t)n * 256 + oi + 1] - 0.5f;
            float aw = attw[(size_t)n * 128 + m * 16 + l * 4 + p];

            float xf = floorf(x), yf = floorf(y);
            int x0 = (int)xf, y0 = (int)yf;
            float wx = x - xf, wy = y - yf;
            float cw[4] = {(1.f - wy) * (1.f - wx), (1.f - wy) * wx,
                           wy * (1.f - wx),         wy * wx};
            int xs[4] = {x0, x0 + 1, x0,     x0 + 1};
            int ys[4] = {y0, y0,     y0 + 1, y0 + 1};
            float s = 0.f;
#pragma unroll
            for (int c = 0; c < 4; ++c) {
                int xi = xs[c], yi = ys[c];
                if (xi >= 0 && xi < W && yi >= 0 && yi < H)
                    s += cw[c] * value[lbase + (size_t)(yi * W + xi) * 256];
            }
            acc += aw * s;
        }
    }
    samp[(size_t)n * 256 + m * 32 + lane] = (_Float16)acc;
}

// ---------------- residual + LayerNorm(256): one wave per row -------------
__global__ __launch_bounds__(256)
void ln_kernel(const float* __restrict__ x1, const float* __restrict__ x2,
               const float* __restrict__ g,  const float* __restrict__ bb,
               float* __restrict__ outf, _Float16* __restrict__ outh,
               int N, int Nrows)
{
    int row  = blockIdx.x * 8 + (threadIdx.x >> 5);
    int lane = threadIdx.x & 31;
    if (row >= Nrows) return;
    size_t base = (size_t)row * 256;
    if (row >= N) {                  // deterministic pad rows
#pragma unroll
        for (int j = 0; j < 8; ++j) {
            int e = j * 32 + lane;
            if (outf) outf[base + e] = 0.f;
            if (outh) outh[base + e] = (_Float16)0.f;
        }
        return;
    }
    float x[8], s = 0.f, sq = 0.f;
#pragma unroll
    for (int j = 0; j < 8; ++j) {
        int e = j * 32 + lane;
        x[j] = x1[base + e] + x2[base + e];
        s += x[j]; sq += x[j] * x[j];
    }
#pragma unroll
    for (int o = 16; o > 0; o >>= 1) {
        s  += __shfl_xor(s,  o, 32);
        sq += __shfl_xor(sq, o, 32);
    }
    float mean = s * (1.f / 256.f);
    float var  = sq * (1.f / 256.f) - mean * mean;
    float rs   = rsqrtf(var + 1e-5f);
#pragma unroll
    for (int j = 0; j < 8; ++j) {
        int e = j * 32 + lane;
        float yv = (x[j] - mean) * rs * g[e] + bb[e];
        if (outf) outf[base + e] = yv;
        if (outh) outh[base + e] = (_Float16)yv;
    }
}

// ---------------- driver ---------------------------------------------------
extern "C" void kernel_launch(void* const* d_in, const int* in_sizes, int n_in,
                              void* d_out, int out_size, void* d_ws, size_t ws_size,
                              hipStream_t stream)
{
    (void)in_sizes; (void)n_in; (void)out_size; (void)ws_size;
    constexpr int Nq = 45012, Npad = 45056, Lseq = 11253;

    const float* src    = (const float*)d_in[0];
    const float* ref    = (const float*)d_in[2];
    const float* q_pos  = (const float*)d_in[3];
    const float* q_feat = (const float*)d_in[4];
    const float* W_off  = (const float*)d_in[5];
    const float* b_off  = (const float*)d_in[6];
    const float* W_attn = (const float*)d_in[7];
    const float* b_attn = (const float*)d_in[8];
    const float* W_v    = (const float*)d_in[9];
    const float* b_v    = (const float*)d_in[10];
    const float* W_out  = (const float*)d_in[11];
    const float* b_out  = (const float*)d_in[12];
    const float* ln1_g  = (const float*)d_in[13];
    const float* ln1_b  = (const float*)d_in[14];
    const float* W1     = (const float*)d_in[15];
    const float* b1     = (const float*)d_in[16];
    const float* W2     = (const float*)d_in[17];
    const float* b2     = (const float*)d_in[18];
    const float* ln2_g  = (const float*)d_in[19];
    const float* ln2_b  = (const float*)d_in[20];

    char* ws = (char*)d_ws;
    size_t o = 0;
    auto alloc = [&](size_t bytes) -> char* {
        char* p = ws + o; o += (bytes + 255) & ~(size_t)255; return p;
    };
    _Float16* src16  = (_Float16*)alloc((size_t)Npad * 256 * 2);
    _Float16* q16    = (_Float16*)alloc((size_t)Npad * 256 * 2);
    _Float16* wv_t   = (_Float16*)alloc((size_t)256 * 256 * 2);
    _Float16* woff_t = (_Float16*)alloc((size_t)256 * 256 * 2);
    _Float16* wattn_t= (_Float16*)alloc((size_t)256 * 128 * 2);
    _Float16* wout_t = (_Float16*)alloc((size_t)256 * 256 * 2);
    _Float16* w1_t   = (_Float16*)alloc((size_t)256 * 1024 * 2);
    _Float16* w2_t   = (_Float16*)alloc((size_t)1024 * 256 * 2);
    float*    value  = (float*)   alloc((size_t)Npad * 256 * 4);
    float*    offb   = (float*)   alloc((size_t)Npad * 256 * 4);
    float*    attn   = (float*)   alloc((size_t)Npad * 128 * 4);
    _Float16* samp16 = (_Float16*)alloc((size_t)Npad * 256 * 2);
    _Float16* ffn1   = (_Float16*)alloc((size_t)Npad * 1024 * 2);
    // buffer reuse (all safe by stream ordering):
    float*    attn_out = offb;    // off consumed by sampling before W_out GEMM
    float*    h_f32    = value;   // value consumed by sampling before LN1
    _Float16* h16      = q16;     // query consumed by off/attn GEMMs before LN1
    float*    ffn2     = offb;    // attn_out consumed by LN1 before FFN2 GEMM

    prep_kernel<<<Npad, 256, 0, stream>>>(src, q_feat, q_pos, src16, q16, samp16, Nq);

    wtrans_kernel<<<(256 * 256 + 255) / 256, 256, 0, stream>>>(W_v,    wv_t,    256, 256);
    wtrans_kernel<<<(256 * 256 + 255) / 256, 256, 0, stream>>>(W_off,  woff_t,  256, 256);
    wtrans_kernel<<<(256 * 128 + 255) / 256, 256, 0, stream>>>(W_attn, wattn_t, 256, 128);
    wtrans_kernel<<<(256 * 256 + 255) / 256, 256, 0, stream>>>(W_out,  wout_t,  256, 256);
    wtrans_kernel<<<(256 * 1024 + 255) / 256, 256, 0, stream>>>(W1,    w1_t,    256, 1024);
    wtrans_kernel<<<(1024 * 256 + 255) / 256, 256, 0, stream>>>(W2,    w2_t,   1024, 256);

    dim3 g256(Npad / 128, 256 / 64);
    wmma_gemm_kernel<<<g256, 256, 0, stream>>>(src16, wv_t, b_v, value, nullptr, 256, 256, 0);
    wmma_gemm_kernel<<<g256, 256, 0, stream>>>(q16, woff_t, b_off, offb, nullptr, 256, 256, 0);
    wmma_gemm_kernel<<<dim3(Npad / 128, 128 / 64), 256, 0, stream>>>(
        q16, wattn_t, b_attn, attn, nullptr, 256, 128, 0);

    softmax_kernel<<<(Nq * 8 + 255) / 256, 256, 0, stream>>>(attn, Nq);
    sample_kernel<<<Nq, 256, 0, stream>>>(value, offb, attn, ref, samp16, Lseq);

    wmma_gemm_kernel<<<g256, 256, 0, stream>>>(samp16, wout_t, b_out, attn_out, nullptr, 256, 256, 0);
    ln_kernel<<<Npad / 8, 256, 0, stream>>>(q_feat, attn_out, ln1_g, ln1_b, h_f32, h16, Nq, Npad);

    wmma_gemm_kernel<<<dim3(Npad / 128, 1024 / 64), 256, 0, stream>>>(
        h16, w1_t, b1, nullptr, ffn1, 256, 1024, 1);
    wmma_gemm_kernel<<<g256, 256, 0, stream>>>(ffn1, w2_t, b2, ffn2, nullptr, 1024, 256, 0);

    ln_kernel<<<(Nq + 7) / 8, 256, 0, stream>>>(h_f32, ffn2, ln2_g, ln2_b,
                                                (float*)d_out, nullptr, Nq, Nq);
}
